// DecoderTransformerLayer_33363305955454
// MI455X (gfx1250) — matrix operand order repaired and checked by reading, not compiled, not run on hardware
//
#include <hip/hip_runtime.h>
#include <hip/hip_bf16.h>
#include <math.h>

// ---------------------------------------------------------------------------
// Types for CDNA5 WMMA (wave32): bf16 inputs, f32 accumulate
// ---------------------------------------------------------------------------
typedef __bf16 bf16x16 __attribute__((ext_vector_type(16)));
typedef __bf16 bf16x8  __attribute__((ext_vector_type(8)));
typedef float  f32x8   __attribute__((ext_vector_type(8)));
typedef int    i32x4vs __attribute__((vector_size(16)));   // builtin's pointee type

// CDNA5 async global->LDS (ASYNCcnt-tracked DMA), guarded by builtin probe.
#if defined(__gfx1250__) && __has_builtin(__builtin_amdgcn_global_load_async_to_lds_b128) && __has_builtin(__builtin_amdgcn_s_wait_asynccnt)
#define USE_ASYNC_LDS 1
#else
#define USE_ASYNC_LDS 0
#endif

#if USE_ASYNC_LDS
__device__ __forceinline__ void async_copy_b128(const __bf16* g, __bf16* l) {
    __builtin_amdgcn_global_load_async_to_lds_b128(
        (__attribute__((address_space(1))) i32x4vs*)(void*)const_cast<__bf16*>(g),
        (__attribute__((address_space(3))) i32x4vs*)(void*)l,
        0, 0);
}
#endif

__device__ __forceinline__ __bf16 f32_to_bf16(float f) {
    union { float f; unsigned u; } v; v.f = f;
    unsigned r = v.u + 0x7FFFu + ((v.u >> 16) & 1u);   // round-to-nearest-even
    unsigned short h = (unsigned short)(r >> 16);
    return __builtin_bit_cast(__bf16, h);
}

__device__ __forceinline__ f32x8 fzero8() {
    f32x8 v;
#pragma unroll
    for (int i = 0; i < 8; ++i) v[i] = 0.0f;
    return v;
}

// A-fragment (16x32 bf16, MxK): lane L(0-15) row M=L holds K=0..7,16..23;
// lanes 16-31 hold K=8..15,24..31  (per CDNA5 ISA 7.12.2)
__device__ __forceinline__ bf16x16 load_a_frag(const __bf16* tile, int ld) {
    int lane = threadIdx.x & 31;
    int m    = lane & 15;
    int half = lane >> 4;
    const __bf16* r = tile + (size_t)m * ld;
    int k0 = half * 8;
    int k1 = 16 + half * 8;
    bf16x16 a;
#pragma unroll
    for (int i = 0; i < 8; ++i) { a[i] = r[k0 + i]; a[8 + i] = r[k1 + i]; }
    return a;
}

// B-fragment (32x16 bf16, KxN) sourced from a [N][K] row-major tile:
// lane L(0-15) column N=L holds K=0..7,16..23; lanes 16-31 hold K=8..15,24..31
__device__ __forceinline__ bf16x16 load_b_frag(const __bf16* tileNT, int ld) {
    int lane = threadIdx.x & 31;
    int n    = lane & 15;
    int half = lane >> 4;
    const __bf16* r = tileNT + (size_t)n * ld + half * 8;
    bf16x16 b;
#pragma unroll
    for (int i = 0; i < 8; ++i) { b[i] = r[i]; b[8 + i] = r[16 + i]; }
    return b;
}

__device__ __forceinline__ f32x8 wmma_bf16(bf16x16 a, bf16x16 b, f32x8 c) {
    return __builtin_amdgcn_wmma_f32_16x16x32_bf16(false, a, false, b, (short)0, c,
                                                   false, false);
}

__device__ __forceinline__ float gelu_exact(float x) {
    return 0.5f * x * (1.0f + erff(x * 0.70710678118654752f));
}

// ---------------------------------------------------------------------------
// fp32 -> bf16 elementwise convert
// ---------------------------------------------------------------------------
__global__ void cvt_f32_bf16(const float* __restrict__ in, __bf16* __restrict__ out,
                             size_t n) {
    size_t i = (size_t)blockIdx.x * blockDim.x + threadIdx.x;
    if (i < n) out[i] = f32_to_bf16(in[i]);
}

// wq[h][d][k] ([H,D,DH]) -> out[(h*DH+k)][d]  ([N=1024][K=1024] row-major, bf16)
__global__ void cvt_qkv_weight(const float* __restrict__ in, __bf16* __restrict__ out,
                               int D, int DH) {
    size_t i = (size_t)blockIdx.x * blockDim.x + threadIdx.x;  // over N*D
    size_t ND = (size_t)D * D;
    if (i >= ND) return;
    int d = (int)(i % D);
    int n = (int)(i / D);
    int h = n / DH;
    int k = n % DH;
    out[i] = f32_to_bf16(in[(size_t)h * D * DH + (size_t)d * DH + k]);
}

// ---------------------------------------------------------------------------
// GEMM: C[M,N] = A[M,K](bf16,row-major) * W[N,K](bf16,row-major)^T
// block tile 128x128, 4 waves (128 thr) of 64x64, BK=32, f32 accum.
// LDS double-buffered; tiles fetched with GLOBAL_LOAD_ASYNC_TO_LDS_B128
// (ASYNCcnt pipeline) when available, else batched register staging.
// Epilogue: optional bias, optional exact GELU, writes f32 and/or bf16.
// ---------------------------------------------------------------------------
#define GBM 128
#define GBN 128
#define GBK 32
#define GTILE (GBM * GBK)

__global__ __launch_bounds__(128)
void gemm_bf16_wmma(const __bf16* __restrict__ A, const __bf16* __restrict__ W,
                    int M, int N, int K, const float* __restrict__ bias,
                    int act, float* __restrict__ outF, __bf16* __restrict__ outB) {
    __shared__ __attribute__((aligned(16))) __bf16 sA[2 * GTILE];
    __shared__ __attribute__((aligned(16))) __bf16 sW[2 * GTILE];

    const int tid  = threadIdx.x;
    const int wave = tid >> 5;
    const int wr   = wave >> 1;          // 0..1
    const int wc   = wave & 1;           // 0..1
    const int m0   = blockIdx.y * GBM;
    const int n0   = blockIdx.x * GBN;

    f32x8 acc[4][4];
#pragma unroll
    for (int mi = 0; mi < 4; ++mi)
#pragma unroll
        for (int ni = 0; ni < 4; ++ni) acc[mi][ni] = fzero8();

    // issue one 128x32 A tile + 128x32 W tile into LDS buffer `buf`
    auto issue_tile = [&](int kt, int buf) {
#if USE_ASYNC_LDS
#pragma unroll
        for (int i = 0; i < 4; ++i) {
            int c   = tid + i * 128;        // 0..511
            int row = c >> 2;
            int col = (c & 3) << 3;
            async_copy_b128(A + (size_t)(m0 + row) * K + kt + col,
                            sA + buf * GTILE + row * GBK + col);
            async_copy_b128(W + (size_t)(n0 + row) * K + kt + col,
                            sW + buf * GTILE + row * GBK + col);
        }
#else
        bf16x8 ta[4], tw[4];
        int rw[4], cl[4];
#pragma unroll
        for (int i = 0; i < 4; ++i) {
            int c = tid + i * 128;
            rw[i] = c >> 2;
            cl[i] = (c & 3) << 3;
            ta[i] = *(const bf16x8*)(A + (size_t)(m0 + rw[i]) * K + kt + cl[i]);
            tw[i] = *(const bf16x8*)(W + (size_t)(n0 + rw[i]) * K + kt + cl[i]);
        }
#pragma unroll
        for (int i = 0; i < 4; ++i) {
            *(bf16x8*)(sA + buf * GTILE + rw[i] * GBK + cl[i]) = ta[i];
            *(bf16x8*)(sW + buf * GTILE + rw[i] * GBK + cl[i]) = tw[i];
        }
#endif
    };

    int cb = 0;
    issue_tile(0, 0);

    for (int kt = 0; kt < K; kt += GBK) {
        if (kt + GBK < K) {
            issue_tile(kt + GBK, cb ^ 1);   // prefetch next tile into other buffer
#if USE_ASYNC_LDS
            __builtin_amdgcn_s_wait_asynccnt(8);   // current tile's 8 DMAs done
#endif
        } else {
#if USE_ASYNC_LDS
            __builtin_amdgcn_s_wait_asynccnt(0);
#endif
        }
        __syncthreads();   // current tile visible to all waves

        const __bf16* cA = sA + cb * GTILE;
        const __bf16* cW = sW + cb * GTILE;
        bf16x16 af[4], bf_[4];
#pragma unroll
        for (int mi = 0; mi < 4; ++mi)
            af[mi] = load_a_frag(cA + (size_t)(wr * 64 + mi * 16) * GBK, GBK);
#pragma unroll
        for (int ni = 0; ni < 4; ++ni)
            bf_[ni] = load_b_frag(cW + (size_t)(wc * 64 + ni * 16) * GBK, GBK);
#pragma unroll
        for (int mi = 0; mi < 4; ++mi)
#pragma unroll
            for (int ni = 0; ni < 4; ++ni)
                acc[mi][ni] = wmma_bf16(af[mi], bf_[ni], acc[mi][ni]);

        __syncthreads();   // done reading buffer cb; it may be overwritten next
        cb ^= 1;
    }

    // epilogue: C element (r, lanehalf) -> M = r + 8*half, N = lane&15
    const int lane = tid & 31;
    const int nl   = lane & 15;
    const int mh   = (lane >> 4) * 8;
#pragma unroll
    for (int mi = 0; mi < 4; ++mi) {
#pragma unroll
        for (int ni = 0; ni < 4; ++ni) {
            int n   = n0 + wc * 64 + ni * 16 + nl;
            float bb = bias ? bias[n] : 0.0f;
#pragma unroll
            for (int r = 0; r < 8; ++r) {
                int m = m0 + wr * 64 + mi * 16 + mh + r;
                float v = acc[mi][ni][r] + bb;
                if (act) v = gelu_exact(v);
                size_t idx = (size_t)m * N + n;
                if (outF) outF[idx] = v;
                if (outB) outB[idx] = f32_to_bf16(v);
            }
        }
    }
}

// ---------------------------------------------------------------------------
// Flash attention (per (b,h)): 64-query blocks, 4 waves x 16 rows, 64-key tiles,
// online softmax, WMMA for QK^T and P*V.  Q/K/V bf16 stored [B*S, H*DH].
// ---------------------------------------------------------------------------
__global__ __launch_bounds__(128)
void flash_attn_wmma(const __bf16* __restrict__ Q, const __bf16* __restrict__ Kg,
                     const __bf16* __restrict__ Vg, __bf16* __restrict__ Ohead,
                     int S, int D, int DH, int H, int causal) {
    const int bh   = blockIdx.y;
    const int b    = bh / H;
    const int h    = bh % H;
    const int qb   = blockIdx.x;
    const int tid  = threadIdx.x;
    const int wave = tid >> 5;
    const int lane = tid & 31;
    const int nl   = lane & 15;
    const int mh   = (lane >> 4) * 8;

    const __bf16* Qp = Q  + (size_t)b * S * D + (size_t)h * DH;
    const __bf16* Kp = Kg + (size_t)b * S * D + (size_t)h * DH;
    const __bf16* Vp = Vg + (size_t)b * S * D + (size_t)h * DH;

    const int q0 = qb * 64 + wave * 16;

    __shared__ __attribute__((aligned(16))) __bf16 sK[64 * 64];
    __shared__ __attribute__((aligned(16))) __bf16 sVt[64 * 64];
    __shared__ __attribute__((aligned(16))) __bf16 sP[4 * 16 * 64];

    // Q fragments for this wave's 16 rows (DH=64 -> two 32-wide k-steps)
    bf16x16 aq[2];
    aq[0] = load_a_frag(Qp + (size_t)q0 * D, D);
    aq[1] = load_a_frag(Qp + (size_t)q0 * D + 32, D);

    f32x8 o[4];
#pragma unroll
    for (int ni = 0; ni < 4; ++ni) o[ni] = fzero8();
    float mrow[8], lrow[8];
#pragma unroll
    for (int r = 0; r < 8; ++r) { mrow[r] = -3.0e38f; lrow[r] = 0.0f; }

    const float scale = 0.125f;  // 1/sqrt(64)
    const int   tend  = causal ? (qb + 1) * 64 : S;

    for (int t0 = 0; t0 < tend; t0 += 64) {
        __syncthreads();  // previous iteration's LDS reads done
        // batched staging: load all K/V chunks, then store (one wait, not 8)
        bf16x8 tk[4], tv[4];
        int rw[4], cl[4];
#pragma unroll
        for (int i = 0; i < 4; ++i) {
            int c = tid + i * 128;          // 0..511
            rw[i] = c >> 3;                 // 0..63 key index
            cl[i] = (c & 7) << 3;           // 0..56
            tk[i] = *(const bf16x8*)(Kp + (size_t)(t0 + rw[i]) * D + cl[i]);
            tv[i] = *(const bf16x8*)(Vp + (size_t)(t0 + rw[i]) * D + cl[i]);
        }
#pragma unroll
        for (int i = 0; i < 4; ++i) {
            *(bf16x8*)(sK + rw[i] * 64 + cl[i]) = tk[i];
#pragma unroll
            for (int e = 0; e < 8; ++e) sVt[(cl[i] + e) * 64 + rw[i]] = tv[i][e];
        }
        __syncthreads();

        // S = Q * K^T  (4 key sub-tiles of 16, 2 k-steps)
        f32x8 sacc[4];
#pragma unroll
        for (int j = 0; j < 4; ++j) sacc[j] = fzero8();
#pragma unroll
        for (int ks = 0; ks < 2; ++ks)
#pragma unroll
            for (int j = 0; j < 4; ++j) {
                bf16x16 bk = load_b_frag(sK + (size_t)(j * 16) * 64 + ks * 32, 64);
                sacc[j] = wmma_bf16(aq[ks], bk, sacc[j]);
            }

        // scale + causal mask
#pragma unroll
        for (int j = 0; j < 4; ++j)
#pragma unroll
            for (int r = 0; r < 8; ++r) {
                float s = sacc[j][r] * scale;
                if (causal) {
                    int t = t0 + j * 16 + nl;
                    int q = q0 + mh + r;
                    if (t > q) s = -3.0e38f;
                }
                sacc[j][r] = s;
            }

        // online softmax: row stats live per (r, lane-half); reduce over 16 lanes
        float mnew[8];
#pragma unroll
        for (int r = 0; r < 8; ++r) {
            float v = fmaxf(fmaxf(sacc[0][r], sacc[1][r]),
                            fmaxf(sacc[2][r], sacc[3][r]));
#pragma unroll
            for (int msk = 1; msk < 16; msk <<= 1) v = fmaxf(v, __shfl_xor(v, msk, 32));
            mnew[r] = fmaxf(mrow[r], v);
        }

        float psum[8];
#pragma unroll
        for (int r = 0; r < 8; ++r) psum[r] = 0.0f;
        __bf16* sPw = sP + (size_t)wave * (16 * 64);
#pragma unroll
        for (int j = 0; j < 4; ++j)
#pragma unroll
            for (int r = 0; r < 8; ++r) {
                float pv = __expf(sacc[j][r] - mnew[r]);
                psum[r] += pv;
                sPw[(mh + r) * 64 + j * 16 + nl] = f32_to_bf16(pv);
            }
#pragma unroll
        for (int r = 0; r < 8; ++r) {
#pragma unroll
            for (int msk = 1; msk < 16; msk <<= 1) psum[r] += __shfl_xor(psum[r], msk, 32);
            float alpha = __expf(mrow[r] - mnew[r]);
            lrow[r] = lrow[r] * alpha + psum[r];
            mrow[r] = mnew[r];
#pragma unroll
            for (int ni = 0; ni < 4; ++ni) o[ni][r] *= alpha;
        }
        __syncthreads();  // sP visible to all lanes of the wave

        // O += P * V   (contraction over 64 keys -> 2 k-steps)
#pragma unroll
        for (int ks = 0; ks < 2; ++ks) {
            bf16x16 ap = load_a_frag(sPw + ks * 32, 64);
#pragma unroll
            for (int ni = 0; ni < 4; ++ni) {
                bf16x16 bv = load_b_frag(sVt + (size_t)(ni * 16) * 64 + ks * 32, 64);
                o[ni] = wmma_bf16(ap, bv, o[ni]);
            }
        }
    }

    // finalize: divide by softmax denominator, write bf16 heads [B*S, H*DH]
#pragma unroll
    for (int ni = 0; ni < 4; ++ni)
#pragma unroll
        for (int r = 0; r < 8; ++r) {
            float v = o[ni][r] / lrow[r];
            size_t idx = (size_t)(b * S + q0 + mh + r) * D + h * DH + ni * 16 + nl;
            Ohead[idx] = f32_to_bf16(v);
        }
}

// ---------------------------------------------------------------------------
// Residual add + LayerNorm over D=1024; writes f32 and optionally bf16
// ---------------------------------------------------------------------------
__global__ __launch_bounds__(256)
void resln_kernel(const float* __restrict__ x, const float* __restrict__ res,
                  const float* __restrict__ g, const float* __restrict__ bvec,
                  float* __restrict__ outF, __bf16* __restrict__ outB) {
    const int D_ = 1024;
    const int row = blockIdx.x;
    __shared__ float vbuf[1024];
    __shared__ float r1[8], r2[8];
    size_t base = (size_t)row * D_;
    float s = 0.0f, s2 = 0.0f;
    for (int i = threadIdx.x; i < D_; i += 256) {
        float v = x[base + i] + (res ? res[base + i] : 0.0f);
        vbuf[i] = v; s += v; s2 += v * v;
    }
#pragma unroll
    for (int msk = 1; msk < 32; msk <<= 1) {
        s += __shfl_xor(s, msk, 32);
        s2 += __shfl_xor(s2, msk, 32);
    }
    int wv = threadIdx.x >> 5, ln = threadIdx.x & 31;
    if (ln == 0) { r1[wv] = s; r2[wv] = s2; }
    __syncthreads();
    s = 0.0f; s2 = 0.0f;
#pragma unroll
    for (int w = 0; w < 8; ++w) { s += r1[w]; s2 += r2[w]; }
    float mu   = s / D_;
    float var  = s2 / D_ - mu * mu;
    float rstd = rsqrtf(var + 1e-5f);
    for (int i = threadIdx.x; i < D_; i += 256) {
        float y = (vbuf[i] - mu) * rstd * g[i] + bvec[i];
        if (outF) outF[base + i] = y;
        if (outB) outB[base + i] = f32_to_bf16(y);
    }
}

// ---------------------------------------------------------------------------
// Host orchestration
// ---------------------------------------------------------------------------
extern "C" void kernel_launch(void* const* d_in, const int* in_sizes, int n_in,
                              void* d_out, int out_size, void* d_ws, size_t ws_size,
                              hipStream_t stream) {
    (void)in_sizes; (void)n_in; (void)out_size; (void)ws_size;
    const float* embeds = (const float*)d_in[0];
    const float* enc    = (const float*)d_in[1];
    const float* wq1 = (const float*)d_in[2];
    const float* wk1 = (const float*)d_in[3];
    const float* wv1 = (const float*)d_in[4];
    const float* wo1 = (const float*)d_in[5];
    const float* wq2 = (const float*)d_in[6];
    const float* wk2 = (const float*)d_in[7];
    const float* wv2 = (const float*)d_in[8];
    const float* wo2 = (const float*)d_in[9];
    const float* ln1g = (const float*)d_in[10];
    const float* ln1b = (const float*)d_in[11];
    const float* ln2g = (const float*)d_in[12];
    const float* ln2b = (const float*)d_in[13];
    const float* ln3g = (const float*)d_in[14];
    const float* ln3b = (const float*)d_in[15];
    const float* w1 = (const float*)d_in[16];
    const float* b1 = (const float*)d_in[17];
    const float* w2 = (const float*)d_in[18];
    const float* b2 = (const float*)d_in[19];

    constexpr int Bb = 8, S = 1024, Dd = 1024, H = 16, DH = 64, F = 4096;
    const size_t MD = (size_t)Bb * S * Dd;   // 8 Mi
    const size_t DD = (size_t)Dd * Dd;       // 1 Mi
    const size_t FD = (size_t)F * Dd;        // 4 Mi

    char* p = (char*)d_ws;
    auto take = [&](size_t bytes) -> char* {
        char* r = p; p += (bytes + 255) & ~(size_t)255; return r;
    };
    __bf16* embB  = (__bf16*)take(MD * 2);
    __bf16* encB  = (__bf16*)take(MD * 2);
    __bf16* wq1t  = (__bf16*)take(DD * 2);
    __bf16* wk1t  = (__bf16*)take(DD * 2);
    __bf16* wv1t  = (__bf16*)take(DD * 2);
    __bf16* wo1B  = (__bf16*)take(DD * 2);
    __bf16* wq2t  = (__bf16*)take(DD * 2);
    __bf16* wk2t  = (__bf16*)take(DD * 2);
    __bf16* wv2t  = (__bf16*)take(DD * 2);
    __bf16* wo2B  = (__bf16*)take(DD * 2);
    __bf16* w1B   = (__bf16*)take(FD * 2);
    __bf16* w2B   = (__bf16*)take(FD * 2);
    __bf16* Qb    = (__bf16*)take(MD * 2);
    __bf16* Kb    = (__bf16*)take(MD * 2);
    __bf16* Vb    = (__bf16*)take(MD * 2);
    __bf16* heads = (__bf16*)take(MD * 2);
    float*  tmpF  = (float*)take(MD * 4);   // sa / ca / ff reuse
    float*  x1F   = (float*)take(MD * 4);
    float*  x2F   = (float*)take(MD * 4);
    __bf16* xB    = (__bf16*)take(MD * 2);  // x1 bf16 then x2 bf16
    __bf16* hmid  = (__bf16*)take((size_t)Bb * S * F * 2);

    const int M = Bb * S;  // 8192
    dim3 blk256(256), blk128(128);

    // --- convert inputs to bf16 ---
    cvt_f32_bf16<<<dim3((unsigned)(MD / 256)), blk256, 0, stream>>>(embeds, embB, MD);
    cvt_f32_bf16<<<dim3((unsigned)(MD / 256)), blk256, 0, stream>>>(enc, encB, MD);
    cvt_f32_bf16<<<dim3((unsigned)(DD / 256)), blk256, 0, stream>>>(wo1, wo1B, DD);
    cvt_f32_bf16<<<dim3((unsigned)(DD / 256)), blk256, 0, stream>>>(wo2, wo2B, DD);
    cvt_f32_bf16<<<dim3((unsigned)(FD / 256)), blk256, 0, stream>>>(w1, w1B, FD);
    cvt_f32_bf16<<<dim3((unsigned)(FD / 256)), blk256, 0, stream>>>(w2, w2B, FD);
    cvt_qkv_weight<<<dim3((unsigned)(DD / 256)), blk256, 0, stream>>>(wq1, wq1t, Dd, DH);
    cvt_qkv_weight<<<dim3((unsigned)(DD / 256)), blk256, 0, stream>>>(wk1, wk1t, Dd, DH);
    cvt_qkv_weight<<<dim3((unsigned)(DD / 256)), blk256, 0, stream>>>(wv1, wv1t, Dd, DH);
    cvt_qkv_weight<<<dim3((unsigned)(DD / 256)), blk256, 0, stream>>>(wq2, wq2t, Dd, DH);
    cvt_qkv_weight<<<dim3((unsigned)(DD / 256)), blk256, 0, stream>>>(wk2, wk2t, Dd, DH);
    cvt_qkv_weight<<<dim3((unsigned)(DD / 256)), blk256, 0, stream>>>(wv2, wv2t, Dd, DH);

    dim3 gD(Dd / GBN, M / GBM);   // N=1024 GEMMs
    dim3 gF(F / GBN, M / GBM);    // N=4096 GEMM
    dim3 gAtt(S / 64, Bb * H);

    // --- 1) causal self-attention ---
    gemm_bf16_wmma<<<gD, blk128, 0, stream>>>(embB, wq1t, M, Dd, Dd, nullptr, 0, nullptr, Qb);
    gemm_bf16_wmma<<<gD, blk128, 0, stream>>>(embB, wk1t, M, Dd, Dd, nullptr, 0, nullptr, Kb);
    gemm_bf16_wmma<<<gD, blk128, 0, stream>>>(embB, wv1t, M, Dd, Dd, nullptr, 0, nullptr, Vb);
    flash_attn_wmma<<<gAtt, blk128, 0, stream>>>(Qb, Kb, Vb, heads, S, Dd, DH, H, 1);
    gemm_bf16_wmma<<<gD, blk128, 0, stream>>>(heads, wo1B, M, Dd, Dd, nullptr, 0, tmpF, nullptr);
    resln_kernel<<<dim3(M), blk256, 0, stream>>>(tmpF, embeds, ln1g, ln1b, x1F, xB);

    // --- 2) cross-attention ---
    gemm_bf16_wmma<<<gD, blk128, 0, stream>>>(xB,   wq2t, M, Dd, Dd, nullptr, 0, nullptr, Qb);
    gemm_bf16_wmma<<<gD, blk128, 0, stream>>>(encB, wk2t, M, Dd, Dd, nullptr, 0, nullptr, Kb);
    gemm_bf16_wmma<<<gD, blk128, 0, stream>>>(encB, wv2t, M, Dd, Dd, nullptr, 0, nullptr, Vb);
    flash_attn_wmma<<<gAtt, blk128, 0, stream>>>(Qb, Kb, Vb, heads, S, Dd, DH, H, 0);
    gemm_bf16_wmma<<<gD, blk128, 0, stream>>>(heads, wo2B, M, Dd, Dd, nullptr, 0, tmpF, nullptr);
    resln_kernel<<<dim3(M), blk256, 0, stream>>>(tmpF, x1F, ln2g, ln2b, x2F, xB);

    // --- 3) FFN: GELU(x2 @ w1^T + b1) @ w2^T + b2 ---
    gemm_bf16_wmma<<<gF, blk128, 0, stream>>>(xB, w1B, M, F, Dd, b1, 1, nullptr, hmid);
    gemm_bf16_wmma<<<gD, blk128, 0, stream>>>(hmid, w2B, M, Dd, F, b2, 0, tmpF, nullptr);
    resln_kernel<<<dim3(M), blk256, 0, stream>>>(tmpF, x2F, ln3g, ln3b, (float*)d_out, nullptr);
}
